// GNN_39075612459051
// MI455X (gfx1250) — matrix operand order repaired and checked by reading, not compiled, not run on hardware
//
#include <hip/hip_runtime.h>
#include <hip/hip_bf16.h>
#include <stdint.h>
#include <math.h>

// Problem constants (from reference setup_inputs)
#define NNODES 100000
#define CH     128
#define NEDGES 800000
#define COUT   47
#define COUTP  48      // padded to 3 WMMA column tiles

typedef __attribute__((ext_vector_type(16))) __bf16 v16bf;
typedef __attribute__((ext_vector_type(8)))  __bf16 v8bf;
typedef __attribute__((ext_vector_type(8)))  float  v8f;

// ---------------------------------------------------------------------------
// Fragment helpers (CDNA5 ISA 7.12.2 layouts, wave32)
// ---------------------------------------------------------------------------

// B (32x16, bf16): lane holds column n = lane&15; two 8-wide K chunks at
// kB = k0 + 8*(lane>=16) and kB+16. Weights stored row-major [n][k] (=W^T use).
__device__ __forceinline__ v16bf load_b_frag(const __bf16* __restrict__ p) {
  const v8bf* q = (const v8bf*)p;          // 16B-aligned (K offsets mult. of 8)
  v8bf lo = q[0], hi = q[2];
  return __builtin_shufflevector(lo, hi, 0,1,2,3,4,5,6,7,8,9,10,11,12,13,14,15);
}

// A (16x32, bf16) from LDS-staged tile: lane holds row m = lane&15; two 8-wide
// K chunks at kA = k0 + 8*(lane>=16) and kA+16. Two ds_load_b128.
__device__ __forceinline__ v16bf lds_a_frag(const __bf16* p0, const __bf16* p1) {
  v8bf lo = *(const v8bf*)p0;
  v8bf hi = *(const v8bf*)p1;
  return __builtin_shufflevector(lo, hi, 0,1,2,3,4,5,6,7,8,9,10,11,12,13,14,15);
}

// ---------------------------------------------------------------------------
// Kernels
// ---------------------------------------------------------------------------

__global__ void __launch_bounds__(256)
zero_f32(float* __restrict__ p, int n) {
  for (int i = blockIdx.x * blockDim.x + threadIdx.x; i < n;
       i += gridDim.x * blockDim.x)
    p[i] = 0.0f;
}

// bf16 weight pre-conversion (+ zero-pad w2 row 47) + zero BN stats slab
__global__ void __launch_bounds__(256)
prep_weights(const float* __restrict__ w1l, const float* __restrict__ w1r,
             const float* __restrict__ w2l, const float* __restrict__ w2r,
             __bf16* __restrict__ w1lb, __bf16* __restrict__ w1rb,
             __bf16* __restrict__ w2lb, __bf16* __restrict__ w2rb,
             float* __restrict__ stats) {
  int t = blockIdx.x * 256 + threadIdx.x;
  if (t < CH * CH) {
    w1lb[t] = (__bf16)w1l[t];
    w1rb[t] = (__bf16)w1r[t];
  }
  if (t < COUTP * CH) {
    int r = t / CH, c = t % CH;
    float a = (r < COUT) ? w2l[r * CH + c] : 0.0f;
    float b = (r < COUT) ? w2r[r * CH + c] : 0.0f;
    w2lb[t] = (__bf16)a;
    w2rb[t] = (__bf16)b;
  }
  if (t < 512) stats[t] = 0.0f;
}

// One wave per edge; lane covers 4 channels. fp32 atomics stay in L2
// (51.2MB feature matrix << 192MB L2). cnt==nullptr on layer 2 (degrees reused).
__global__ void __launch_bounds__(256)
sage_scatter(const float* __restrict__ feat, const long long* __restrict__ ei,
             float* __restrict__ agg, float* __restrict__ cnt, int nedges) {
  int g = blockIdx.x * blockDim.x + threadIdx.x;
  int e = g >> 5, lane = g & 31;
  if (e >= nedges) return;
  long long s = ei[e];
  long long d = ei[(size_t)nedges + e];
  const float4 v = *(const float4*)(feat + (size_t)s * CH + lane * 4);
  float* dp = agg + (size_t)d * CH + lane * 4;
  atomicAdd(dp + 0, v.x);
  atomicAdd(dp + 1, v.y);
  atomicAdd(dp + 2, v.z);
  atomicAdd(dp + 3, v.w);
  if (cnt != nullptr && lane == 0) atomicAdd(cnt + d, 1.0f);
}

// Layer 1: out = mean_agg @ wl^T + b + x @ wr^T, then row-L2 normalize.
// Block = 8 waves; block owns 16 nodes; wave w owns columns [16w,16w+16).
// A tiles (scaled agg, x) are converted to bf16 ONCE into LDS cooperatively,
// then all 8 waves read fragments via ds_load_b128 (8x less cvt/global work).
__global__ void __launch_bounds__(256)
sage_gemm1(const float* __restrict__ agg, const float* __restrict__ cnt,
           const float* __restrict__ x, const __bf16* __restrict__ wl,
           const __bf16* __restrict__ wr, const float* __restrict__ bias,
           float* __restrict__ hout) {
  __shared__ __bf16 aT[2][16][CH];     // [0]=mean-agg, [1]=x   (8KB)
  __shared__ float  tile[16][CH];      // f32 epilogue tile     (8KB)
  __shared__ float  nrm[16];
  const int tid  = threadIdx.x;
  const int wave = tid >> 5, lane = tid & 31;
  const int mloc = lane & 15;
  const int kh   = (lane >> 4) * 8;    // A K-half
  const int kbs  = (lane >> 4) * 8;    // B K-half
  const int node0 = blockIdx.x * 16;   // NNODES%16==0 -> no guard
  if (tid < 16) nrm[tid] = 0.0f;

  { // cooperative stage: 2048 elems/matrix, 8 contiguous per thread
    const int r = tid >> 4;            // 0..15
    const int c = (tid & 15) * 8;      // 0..120
    const float sc = 1.0f / fmaxf(cnt[node0 + r], 1.0f);
    const float* ga = agg + (size_t)(node0 + r) * CH + c;
    const float* gx = x   + (size_t)(node0 + r) * CH + c;
#pragma unroll
    for (int j = 0; j < 8; ++j) {
      aT[0][r][c + j] = (__bf16)(ga[j] * sc);
      aT[1][r][c + j] = (__bf16)gx[j];
    }
  }
  __syncthreads();

  const int n = wave * 16 + mloc;      // B column / D column / bias index
  v8f acc = {};
#pragma unroll
  for (int k0 = 0; k0 < CH; k0 += 32) {
    const int kA = k0 + kh;
    v16bf a1 = lds_a_frag(&aT[0][mloc][kA], &aT[0][mloc][kA + 16]);
    v16bf b1 = load_b_frag(wl + (size_t)n * CH + k0 + kbs);
    acc = __builtin_amdgcn_wmma_f32_16x16x32_bf16(false, a1, false, b1,
                                                  (short)0, acc, false, false);
    v16bf a2 = lds_a_frag(&aT[1][mloc][kA], &aT[1][mloc][kA + 16]);
    v16bf b2 = load_b_frag(wr + (size_t)n * CH + k0 + kbs);
    acc = __builtin_amdgcn_wmma_f32_16x16x32_bf16(false, a2, false, b2,
                                                  (short)0, acc, false, false);
  }

  const float bn = bias[n];
  const int rbase = (lane >> 4) * 8;   // D rows: lanes<16 -> 0..7, else 8..15
#pragma unroll
  for (int j = 0; j < 8; ++j)
    tile[rbase + j][n] = acc[j] + bn;
  __syncthreads();

  { // row sum of squares (16 threads x 8 cols each per row)
    const int r = tid >> 4;
    const int cb = (tid & 15) * 8;
    float s = 0.0f;
#pragma unroll
    for (int j = 0; j < 8; ++j) { float v = tile[r][cb + j]; s += v * v; }
    atomicAdd(&nrm[r], s);
  }
  __syncthreads();

#pragma unroll
  for (int it = 0; it < 8; ++it) {     // 2048 elems / 256 threads
    int idx = tid + it * 256;
    int r = idx >> 7, c = idx & (CH - 1);
    float inv = 1.0f / fmaxf(sqrtf(nrm[r]), 1e-12f);
    hout[(size_t)(node0 + r) * CH + c] = tile[r][c] * inv;
  }
}

// BatchNorm pass 1: per-channel sum / sumsq (block covers 256 rows)
__global__ void __launch_bounds__(128)
bn_stats(const float* __restrict__ h, float* __restrict__ stats, int nrows) {
  const int col = threadIdx.x;
  const int r0 = blockIdx.x * 256;
  float s = 0.0f, q = 0.0f;
  for (int r = 0; r < 256; ++r) {
    int rr = r0 + r;
    if (rr >= nrows) break;
    float v = h[(size_t)rr * CH + col];
    s += v; q += v * v;
  }
  atomicAdd(&stats[col], s);
  atomicAdd(&stats[CH + col], q);
}

__global__ void __launch_bounds__(128)
bn_final(float* __restrict__ stats, const float* __restrict__ gamma,
         const float* __restrict__ beta) {
  const int c = threadIdx.x;
  float mean = stats[c] * (1.0f / (float)NNODES);
  float var  = stats[CH + c] * (1.0f / (float)NNODES) - mean * mean;
  float sc   = gamma[c] * rsqrtf(var + 1e-5f);
  stats[256 + c] = sc;
  stats[384 + c] = beta[c] - mean * sc;
}

__global__ void __launch_bounds__(256)
bn_relu(float* __restrict__ h, const float* __restrict__ stats, int total) {
  for (int i = blockIdx.x * blockDim.x + threadIdx.x; i < total;
       i += gridDim.x * blockDim.x) {
    int c = i & (CH - 1);
    float v = h[i] * stats[256 + c] + stats[384 + c];
    h[i] = v > 0.0f ? v : 0.0f;
  }
}

// Layer 2: 16 nodes x 48 (padded) cols; 4 waves, waves 0..2 run WMMA.
// Same LDS-staged bf16 A tiles as layer 1.
__global__ void __launch_bounds__(128)
sage_gemm2(const float* __restrict__ agg, const float* __restrict__ cnt,
           const float* __restrict__ h, const __bf16* __restrict__ wl,
           const __bf16* __restrict__ wr, const float* __restrict__ bias,
           float* __restrict__ out) {
  __shared__ __bf16 aT[2][16][CH];     // 8KB
  __shared__ float  tile[16][COUTP];
  __shared__ float  nrm[16];
  const int tid  = threadIdx.x;
  const int wave = tid >> 5, lane = tid & 31;
  const int mloc = lane & 15;
  const int node0 = blockIdx.x * 16;
  if (tid < 16) nrm[tid] = 0.0f;

  { // cooperative stage with 128 threads: 16 per thread, 8 contiguous in rows r and r+8
#pragma unroll
    for (int half = 0; half < 2; ++half) {
      const int e = tid + half * 128;  // 0..255 "virtual thread"
      const int r = e >> 4;
      const int c = (e & 15) * 8;
      const float sc = 1.0f / fmaxf(cnt[node0 + r], 1.0f);
      const float* ga = agg + (size_t)(node0 + r) * CH + c;
      const float* gh = h   + (size_t)(node0 + r) * CH + c;
#pragma unroll
      for (int j = 0; j < 8; ++j) {
        aT[0][r][c + j] = (__bf16)(ga[j] * sc);
        aT[1][r][c + j] = (__bf16)gh[j];
      }
    }
  }
  __syncthreads();

  if (wave < 3) {                       // wave-uniform: EXEC stays all-ones
    const int kh  = (lane >> 4) * 8;
    const int kbs = (lane >> 4) * 8;
    const int n = wave * 16 + mloc;     // 0..47 (row 47 of padded W is zero)

    v8f acc = {};
#pragma unroll
    for (int k0 = 0; k0 < CH; k0 += 32) {
      const int kA = k0 + kh;
      v16bf a1 = lds_a_frag(&aT[0][mloc][kA], &aT[0][mloc][kA + 16]);
      v16bf b1 = load_b_frag(wl + (size_t)n * CH + k0 + kbs);
      acc = __builtin_amdgcn_wmma_f32_16x16x32_bf16(false, a1, false, b1,
                                                    (short)0, acc, false, false);
      v16bf a2 = lds_a_frag(&aT[1][mloc][kA], &aT[1][mloc][kA + 16]);
      v16bf b2 = load_b_frag(wr + (size_t)n * CH + k0 + kbs);
      acc = __builtin_amdgcn_wmma_f32_16x16x32_bf16(false, a2, false, b2,
                                                    (short)0, acc, false, false);
    }
    const float bn = (n < COUT) ? bias[n] : 0.0f;
    const int rbase = (lane >> 4) * 8;
#pragma unroll
    for (int j = 0; j < 8; ++j)
      tile[rbase + j][n] = acc[j] + bn;
  }
  __syncthreads();

  { // sumsq over 48 cols (col 47 contributes exactly 0)
    const int r = tid >> 3;
    const int cb = (tid & 7) * 6;
    float s = 0.0f;
#pragma unroll
    for (int j = 0; j < 6; ++j) { float v = tile[r][cb + j]; s += v * v; }
    atomicAdd(&nrm[r], s);
  }
  __syncthreads();

  for (int idx = tid; idx < 16 * COUT; idx += 128) {
    int r = idx / COUT, c = idx % COUT;
    float inv = 1.0f / fmaxf(sqrtf(nrm[r]), 1e-12f);
    out[(size_t)(node0 + r) * COUT + c] = tile[r][c] * inv;
  }
}

// ---------------------------------------------------------------------------
// Host launcher
// ---------------------------------------------------------------------------
extern "C" void kernel_launch(void* const* d_in, const int* in_sizes, int n_in,
                              void* d_out, int out_size, void* d_ws, size_t ws_size,
                              hipStream_t stream) {
  const float*     x     = (const float*)d_in[0];
  const long long* ei    = (const long long*)d_in[1];   // int64 edge_index [2,E]
  const float*     w1l   = (const float*)d_in[2];
  const float*     b1l   = (const float*)d_in[3];
  const float*     w1r   = (const float*)d_in[4];
  const float*     gamma = (const float*)d_in[5];
  const float*     beta  = (const float*)d_in[6];
  const float*     w2l   = (const float*)d_in[7];
  const float*     b2l   = (const float*)d_in[8];
  const float*     w2r   = (const float*)d_in[9];
  float* out = (float*)d_out;

  // Workspace layout (~103 MB): agg | h | cnt | stats | bf16 weights
  float*  buf0  = (float*)d_ws;                       // agg       N*CH
  float*  buf1  = buf0 + (size_t)NNODES * CH;         // h         N*CH
  float*  cnt   = buf1 + (size_t)NNODES * CH;         // degrees   N
  float*  stats = cnt + NNODES;                       // 512 f32
  __bf16* w1lb  = (__bf16*)(stats + 512);             // 16B-aligned
  __bf16* w1rb  = w1lb + CH * CH;
  __bf16* w2lb  = w1rb + CH * CH;
  __bf16* w2rb  = w2lb + COUTP * CH;

  const int scatterBlocks = (NEDGES * 32 + 255) / 256;  // 1 wave / edge
  const int nodeBlocks    = NNODES / 16;                // exact (100000/16)

  prep_weights<<<(CH * CH + 255) / 256, 256, 0, stream>>>(
      w1l, w1r, w2l, w2r, w1lb, w1rb, w2lb, w2rb, stats);

  // ---- layer 1 ----
  zero_f32<<<2048, 256, 0, stream>>>(buf0, NNODES * CH);
  zero_f32<<<256, 256, 0, stream>>>(cnt, NNODES);
  sage_scatter<<<scatterBlocks, 256, 0, stream>>>(x, ei, buf0, cnt, NEDGES);
  sage_gemm1<<<nodeBlocks, 256, 0, stream>>>(buf0, cnt, x, w1lb, w1rb, b1l, buf1);

  // ---- batchnorm + relu (in place on h) ----
  bn_stats<<<(NNODES + 255) / 256, 128, 0, stream>>>(buf1, stats, NNODES);
  bn_final<<<1, 128, 0, stream>>>(stats, gamma, beta);
  bn_relu<<<2048, 256, 0, stream>>>(buf1, stats, NNODES * CH);

  // ---- layer 2 (reuse degrees) ----
  zero_f32<<<2048, 256, 0, stream>>>(buf0, NNODES * CH);
  sage_scatter<<<scatterBlocks, 256, 0, stream>>>(buf1, ei, buf0, nullptr, NEDGES);
  sage_gemm2<<<nodeBlocks, 128, 0, stream>>>(buf0, cnt, buf1, w2lb, w2rb, b2l, out);
}
